// ReservoirComputer_12876311953916
// MI455X (gfx1250) — matrix operand order repaired and checked by reading, not compiled, not run on hardware
//
#include <hip/hip_runtime.h>

// ---------------------------------------------------------------------------
// ReservoirComputer on MI455X (gfx1250, wave32, WMMA bf16)
//   h_{t} = 0.1*h_{t-1} + 0.9*tanh( [x_t ; h_{t-1}] @ [W_in ; W] )
//   y_t   = h_t @ R^T + b
// Persistent kernel: 16 owner WGs (64 hidden cols each) + 1 readout WG,
// device-scope barrier per step, h double-buffered in bf16 in workspace.
// ---------------------------------------------------------------------------

#define S_LEN   4096
#define DIN     256
#define DH      1024
#define DOUT    64
#define BATCH   32
#define KTOT    (DIN + DH)      // 1280
#define NWG_OWN 16
#define NWG_TOT (NWG_OWN + 1)   // +1 readout workgroup
#define TPB     256             // 8 waves of 32

typedef unsigned short u16;
typedef __attribute__((ext_vector_type(16))) __bf16 v16bf;
typedef __attribute__((ext_vector_type(8)))  float  v8f;

// f32 -> bf16 round-to-nearest-even
__device__ __forceinline__ u16 f2bf(float f) {
  unsigned u = __float_as_uint(f);
  unsigned r = (u + 0x7FFFu + ((u >> 16) & 1u)) >> 16;
  return (u16)r;
}

// One wave-held bf16 fragment (16 elements / lane = 8 VGPRs)
struct FragBF {
  union { v16bf v; uint4 q[2]; };
};

// A-matrix 16x32 bf16 fragment per ISA 7.12.2:
//   lane L: row M = L%16 ; half = L/16
//   elements 0..7  -> K = kb + half*8      .. +8
//   elements 8..15 -> K = kb + 16 + half*8 .. +8
__device__ __forceinline__ void load_a(FragBF& f, const u16* base, int row,
                                       int stride, int kb, int half) {
  const u16* p = base + (size_t)row * stride + kb + half * 8;
  f.q[0] = *(const uint4*)(p);
  f.q[1] = *(const uint4*)(p + 16);
}

// B-matrix 32x16 bf16 fragment (column-contiguous source = W^T row):
//   lane L: col N = L%16 ; half = L/16 ; elements 0..7 -> K = kb + half*8 + e, 8..15 -> K = kb + 16 + half*8 + (e-8)
__device__ __forceinline__ void load_b(FragBF& f, const u16* colptr, int kb,
                                       int half) {
  const u16* p = colptr + kb + half * 8;
  f.q[0] = *(const uint4*)(p);
  f.q[1] = *(const uint4*)(p + 16);
}

__device__ __forceinline__ v8f wmma_bf16(const FragBF& a, const FragBF& b, v8f c) {
  // D = A(16x32) * B(32x16) + C ; f32 accumulate
  return __builtin_amdgcn_wmma_f32_16x16x32_bf16(
      /*neg_a=*/false, a.v, /*neg_b=*/false, b.v,
      /*c_mod=*/(short)0, c, /*reuse_a=*/false, /*reuse_b=*/false);
}

// Monotonic-counter grid barrier (agent scope). Counter zeroed per launch.
__device__ __forceinline__ void grid_barrier(unsigned* cnt, int t) {
  __threadfence();            // release this block's global writes (agent scope)
  __syncthreads();
  if (threadIdx.x == 0) {
    __hip_atomic_fetch_add(cnt, 1u, __ATOMIC_RELEASE, __HIP_MEMORY_SCOPE_AGENT);
    unsigned tgt = (unsigned)(t + 1) * NWG_TOT;
    while (__hip_atomic_load(cnt, __ATOMIC_ACQUIRE, __HIP_MEMORY_SCOPE_AGENT) < tgt) {
      __builtin_amdgcn_s_sleep(2);
    }
  }
  __syncthreads();
  __threadfence();            // acquire side for all lanes
}

// --------------------------- setup kernels ---------------------------------

__global__ void init_kernel(unsigned* p, unsigned nwords) {
  for (unsigned i = blockIdx.x * blockDim.x + threadIdx.x; i < nwords;
       i += gridDim.x * blockDim.x)
    p[i] = 0u;
}

// WcatT[col][k] = (k < DIN ? W_in[k][col] : W[k-DIN][col])  (bf16, col-major slice)
// RT[o][k]     = readout_w[o][k]                            (bf16)
__global__ void pack_kernel(const float* __restrict__ W_in,
                            const float* __restrict__ W,
                            const float* __restrict__ Rw,
                            u16* __restrict__ WcatT, u16* __restrict__ RT) {
  const size_t N1 = (size_t)DH * KTOT;
  const size_t N2 = (size_t)DOUT * DH;
  size_t stride = (size_t)gridDim.x * blockDim.x;
  for (size_t i = blockIdx.x * (size_t)blockDim.x + threadIdx.x; i < N1 + N2;
       i += stride) {
    if (i < N1) {
      size_t col = i / KTOT, k = i % KTOT;
      float v = (k < DIN) ? W_in[k * DH + col] : W[(k - DIN) * DH + col];
      WcatT[col * KTOT + k] = f2bf(v);
    } else {
      size_t j = i - N1;
      RT[j] = f2bf(Rw[j]);
    }
  }
}

// --------------------------- persistent kernel -----------------------------

__global__ __launch_bounds__(TPB) void reservoir_step_kernel(
    const float* __restrict__ x,        // [B, S, DIN] f32
    const float* __restrict__ rb,       // [DOUT] f32
    const u16* __restrict__ WcatT,      // [DH cols][KTOT] bf16
    const u16* __restrict__ RT,         // [DOUT cols][DH] bf16
    u16* __restrict__ hb0, u16* __restrict__ hb1,  // [B, DH] bf16 ping-pong
    float* __restrict__ y,              // [B, S, DOUT] f32
    unsigned* __restrict__ cnt) {
  __shared__ u16 xs[BATCH * DIN];       // 16 KB: x_t staged as bf16

  const int tid  = threadIdx.x;
  const int wave = tid >> 5;
  const int lane = tid & 31;
  const int nl   = lane & 15;           // A-row / B-col / D-col index
  const int half = lane >> 4;
  const int gid  = blockIdx.x;
  const bool owner = (gid < NWG_OWN);

  const int mr   = (wave & 1) * 16;     // batch-row base of this wave's subtile
  const int ncl  = (wave >> 1) * 16;    // col base within the 64-wide tile
  const int gcol = gid * 64 + ncl;      // owner: global hidden column base

  v8f hreg;                             // owner: persistent f32 h subtile
  for (int j = 0; j < 8; ++j) hreg[j] = 0.0f;

#pragma unroll 1
  for (int t = 0; t < S_LEN; ++t) {
    u16* hin  = (t & 1) ? hb1 : hb0;    // h_{t-1}
    u16* hout = (t & 1) ? hb0 : hb1;    // h_t

    if (owner) {
      // ---- stage x_t -> LDS bf16 ----
      for (int i = tid; i < BATCH * DIN; i += TPB) {
        int b = i >> 8, d = i & (DIN - 1);
        xs[i] = f2bf(x[(size_t)b * S_LEN * DIN + (size_t)t * DIN + d]);
      }
      __syncthreads();

      // ---- z = [x_t ; h] @ Wcat[:, gcol..gcol+15] ----
      v8f acc;
      for (int j = 0; j < 8; ++j) acc[j] = 0.0f;
      const u16* bcol = WcatT + (size_t)(gcol + nl) * KTOT;
#pragma unroll 8
      for (int kb = 0; kb < DIN; kb += 32) {          // x contribution (LDS A)
        FragBF a, b;
        load_a(a, xs, mr + nl, DIN, kb, half);
        load_b(b, bcol, kb, half);
        acc = wmma_bf16(a, b, acc);
      }
#pragma unroll 8
      for (int kb = 0; kb < DH; kb += 32) {           // h contribution (L2 A)
        FragBF a, b;
        load_a(a, hin, mr + nl, DH, kb, half);
        load_b(b, bcol, DIN + kb, half);
        acc = wmma_bf16(a, b, acc);
      }

      // ---- h_t = 0.1*h_{t-1} + 0.9*tanh(z); write bf16 ----
#pragma unroll
      for (int j = 0; j < 8; ++j) {
        float hn = 0.1f * hreg[j] + 0.9f * tanhf(acc[j]);
        hreg[j] = hn;
        int row = mr + half * 8 + j;                  // C/D layout: M = j + 8*half
        hout[(size_t)row * DH + gcol + nl] = f2bf(hn);
      }
    }

    grid_barrier(cnt, t);

    if (!owner) {
      // ---- readout WG (runs in the shadow of the owners' next step):
      //      y_t = h_t @ R^T + b ----
      v8f acc;
      float bias = rb[ncl + nl];
      for (int j = 0; j < 8; ++j) acc[j] = bias;
      const u16* bcol = RT + (size_t)(ncl + nl) * DH;
#pragma unroll 8
      for (int kb = 0; kb < DH; kb += 32) {
        FragBF a, b;
        load_a(a, hout, mr + nl, DH, kb, half);
        load_b(b, bcol, kb, half);
        acc = wmma_bf16(a, b, acc);
      }
#pragma unroll
      for (int j = 0; j < 8; ++j) {
        int row = mr + half * 8 + j;
        y[(size_t)row * S_LEN * DOUT + (size_t)t * DOUT + ncl + nl] = acc[j];
      }
    }
  }
}

// --------------------------- launch ----------------------------------------

extern "C" void kernel_launch(void* const* d_in, const int* in_sizes, int n_in,
                              void* d_out, int out_size, void* d_ws, size_t ws_size,
                              hipStream_t stream) {
  const float* x    = (const float*)d_in[0];  // [32, 4096, 256]
  const float* W_in = (const float*)d_in[1];  // [256, 1024]
  const float* W    = (const float*)d_in[2];  // [1024, 1024]
  const float* Rw   = (const float*)d_in[3];  // [64, 1024]
  const float* rb   = (const float*)d_in[4];  // [64]
  float* y = (float*)d_out;                   // [32, 4096, 64]

  // Workspace carve-up (~2.9 MB total):
  //   [0,256)                 : barrier counter
  //   [256, 256+128K)         : h ping-pong, 2 x [32,1024] bf16
  //   then WcatT [1024][1280] bf16 (2.5 MB), RT [64][1024] bf16 (128 KB)
  char* ws = (char*)d_ws;
  unsigned* cnt = (unsigned*)ws;
  u16* hb0   = (u16*)(ws + 256);
  u16* hb1   = hb0 + (size_t)BATCH * DH;
  u16* WcatT = (u16*)(ws + 256 + 2 * (size_t)BATCH * DH * sizeof(u16));
  u16* RT    = WcatT + (size_t)DH * KTOT;

  // 1) zero barrier counter + h buffers (deterministic across graph replays)
  unsigned zero_words = (256u + 2u * BATCH * DH * sizeof(u16)) / 4u;
  init_kernel<<<64, 256, 0, stream>>>(cnt, zero_words);

  // 2) pack bf16 weights (Wcat transposed for column-contiguous B fragments)
  pack_kernel<<<1024, 256, 0, stream>>>(W_in, W, Rw, WcatT, RT);

  // 3) persistent recurrent scan: 16 owner WGs + 1 readout WG
  reservoir_step_kernel<<<NWG_TOT, TPB, 0, stream>>>(x, rb, WcatT, RT, hb0, hb1,
                                                     y, cnt);
}
